// ARCGridGNNEncoder_78821239816654
// MI455X (gfx1250) — matrix operand order repaired and checked by reading, not compiled, not run on hardware
//
#include <hip/hip_runtime.h>
#include <hip/hip_bf16.h>
#include <hip/hip_fp16.h>

typedef __attribute__((ext_vector_type(16))) _Float16 v16h;
typedef __attribute__((ext_vector_type(8)))  _Float16 v8h;
typedef __attribute__((ext_vector_type(8)))  float    v8f;

#define HID      128
#define NNODES   900
#define WR       30          // grid width/height
#define XSTRIDE  136         // halves per x row (128 + 8 pad -> 68 dwords, bank-friendly)
#define XROWS    912         // 900 nodes padded to 57*16
#define HSTRIDE  136         // halves per h row-node
#define NSLOTS   4           // rolling h-row slots (4 => no end-of-row barrier needed)
#define NLAYERS  4
#define OUTD     64

#define X_HALFS   (XROWS * XSTRIDE)                  // 124032
#define HB_HALFS  (NSLOTS * 32 * HSTRIDE)            // 17408
#define WB_HALFS  (4 * 8 * 32 * 16)                  // 16384
#define SMEM_BYTES ((X_HALFS + HB_HALFS + WB_HALFS) * 2 + (128 * 3 + 64) * 4)

__device__ __forceinline__ float dinv_of(int r, int c) {
    int deg = 1 + (r > 0) + (r < WR - 1) + (c > 0) + (c < WR - 1);
    return rsqrtf((float)deg);
}

// One row (30 nodes, padded to 32) of h = x @ W via v_wmma_f32_16x16x32_f16.
// Each wave owns exactly one channel tile (ct == wave) and both node tiles.
__device__ __forceinline__ void gemm_row(const _Float16* __restrict__ xs,
                                         const _Float16* __restrict__ wbL,
                                         _Float16* __restrict__ hslot,
                                         int nodeBase, int wave, int lane) {
    const int gl = lane >> 4;
    const int ml = lane & 15;
    const int ct = wave;                              // 0..7, wave-uniform
    v16h B[4];
#pragma unroll
    for (int s = 0; s < 4; ++s)
        B[s] = *(const v16h*)(wbL + (size_t)(((s * 8 + ct) * 32) + lane) * 16);
#pragma unroll
    for (int nt = 0; nt < 2; ++nt) {
        v8f acc = {};
        const _Float16* arow = xs + (size_t)(nodeBase + nt * 16 + ml) * XSTRIDE;
#pragma unroll
        for (int s = 0; s < 4; ++s) {
            v8h a0 = *(const v8h*)(arow + 32 * s + 8 * gl);
            v8h a1 = *(const v8h*)(arow + 32 * s + 16 + 8 * gl);
            v16h A = __builtin_shufflevector(a0, a1, 0,1,2,3,4,5,6,7,8,9,10,11,12,13,14,15);
            acc = __builtin_amdgcn_wmma_f32_16x16x32_f16(false, A, false, B[s],
                                                         (short)0, acc, false, false);
        }
#pragma unroll
        for (int v = 0; v < 8; ++v) {                 // D: M = v + 8*gl, N = ml
            int nn = nt * 16 + v + 8 * gl;            // node within row (30,31 = pad)
            hslot[nn * HSTRIDE + ct * 16 + ml] = (_Float16)acc[v];
        }
    }
}

__launch_bounds__(256, 1)
__global__ void arc_gnn_encoder_kernel(const int* __restrict__ grids,
                                       const float* __restrict__ Win,
                                       const float* __restrict__ bin_,
                                       const float* __restrict__ Wg,
                                       const float* __restrict__ bgp,
                                       const float* __restrict__ gammap,
                                       const float* __restrict__ betap,
                                       const float* __restrict__ Wout,
                                       const float* __restrict__ bout,
                                       float* __restrict__ out) {
    extern __shared__ __align__(16) char smem[];
    _Float16* xs = (_Float16*)smem;                   // residual stream, f16
    _Float16* hb = xs + X_HALFS;                      // 4-slot rolling h rows, f16
    _Float16* wb = hb + HB_HALFS;                     // swizzled weights (B operands)
    float* lnG = (float*)(wb + WB_HALFS);             // gamma[128]
    float* lnB = lnG + HID;                           // beta[128]
    float* lnb = lnB + HID;                           // bg[128]
    float* bo  = lnb + HID;                           // bout[64]

    const int tid  = threadIdx.x;
    const int lane = tid & 31;
    const int wave = tid >> 5;
    const int gl   = lane >> 4;
    const int ml   = lane & 15;
    const int gidx = blockIdx.x;

    // ---------------- encode: stage Win/bin/colors in LDS (wb reused as scratch) ----
    {
        float* winS = (float*)wb;                     // 12*128 floats
        float* binS = winS + 12 * HID;                // 128 floats
        int*   colS = (int*)(binS + HID);             // 900 ints  (total 10.3 KB)
        const int* grid = grids + (size_t)gidx * NNODES;
        for (int i = tid; i < 13 * HID; i += 256)
            ((float*)winS)[i] = (i < 12 * HID) ? Win[i] : bin_[i - 12 * HID];
        for (int n = tid; n < NNODES; n += 256) colS[n] = grid[n];
        __syncthreads();

        for (int idx = tid; idx < XROWS * HID; idx += 256) {
            int n  = idx >> 7;
            int ch = idx & 127;
            float v = 0.f;
            if (n < NNODES) {
                int r = n / WR, c = n - r * WR;
                int col = colS[n];
                float f = winS[col * HID + ch]
                        + ((float)r * (1.f / (WR - 1))) * winS[10 * HID + ch]
                        + ((float)c * (1.f / (WR - 1))) * winS[11 * HID + ch]
                        + binS[ch];
                v = fmaxf(f, 0.f);
            }
            xs[(size_t)n * XSTRIDE + ch] = (_Float16)v;   // pad rows -> 0
        }
    }

    // ---------------- GCN layers ----------------
    for (int L = 0; L < NLAYERS; ++L) {
        __syncthreads();   // protect wb / LN params against previous stage's readers

        // swizzle Wg[L] (K x N, f32) into per-lane WMMA-B operand order, f16
        const float* W = Wg + (size_t)L * HID * HID;
        if (L + 1 < NLAYERS)   // prefetch next layer's weights into L2/near caches
            __builtin_prefetch(Wg + (size_t)(L + 1) * HID * HID + tid * 64, 0, 1);
        for (int idx = tid; idx < 4 * 8 * 32; idx += 256) {
            int s  = idx >> 8;
            int l  = idx & 31;
            int gg = l >> 4, mm = l & 15;
            int N  = ((idx >> 5) & 7) * 16 + mm;
            _Float16* dst = wb + (size_t)idx * 16;
#pragma unroll
            for (int e = 0; e < 16; ++e) {
                int K = 32 * s + ((e < 8) ? 8 * gg + e : 16 + 8 * gg + (e - 8));
                dst[e] = (_Float16)W[K * HID + N];
            }
        }
        for (int ch = tid; ch < HID; ch += 256) {
            lnG[ch] = gammap[L * HID + ch];
            lnB[ch] = betap[L * HID + ch];
            lnb[ch] = bgp[L * HID + ch];
        }
        __syncthreads();

        gemm_row(xs, wb, hb + 0, 0, wave, lane);      // prologue: h row 0 -> slot 0
        __syncthreads();

        for (int r = 0; r < WR; ++r) {
            if (r + 1 < WR)                           // h row r+1 -> slot (r+1)&3
                gemm_row(xs, wb, hb + (size_t)((r + 1) & 3) * 32 * HSTRIDE,
                         (r + 1) * WR, wave, lane);
            __syncthreads();                          // only barrier in the row loop

            // ---- stencil + LN (wave-local shuffle reduce) + relu + residual ----
            const int c  = tid >> 3;                  // node column 0..31 (30,31 idle)
            const int cg = tid & 7;                   // channel group: 8 lanes/node
            const int chb = cg * 16;
            if (c < WR) {
                float acc[16];
                const _Float16* h0 = hb + (size_t)(r & 3) * 32 * HSTRIDE;
                const _Float16* hm = hb + (size_t)((r + 3) & 3) * 32 * HSTRIDE;
                const _Float16* hp = hb + (size_t)((r + 1) & 3) * 32 * HSTRIDE;
                const float dC = dinv_of(r, c);
                {   // self loop
                    const v8h* p = (const v8h*)(h0 + c * HSTRIDE + chb);
                    v8h u0 = p[0], u1 = p[1];
                    float w = dC * dC;
#pragma unroll
                    for (int j = 0; j < 8; ++j) { acc[j] = w * (float)u0[j]; acc[8 + j] = w * (float)u1[j]; }
                }
                auto accum = [&](const _Float16* src, float w) {
                    const v8h* p = (const v8h*)src;
                    v8h u0 = p[0], u1 = p[1];
#pragma unroll
                    for (int j = 0; j < 8; ++j) { acc[j] += w * (float)u0[j]; acc[8 + j] += w * (float)u1[j]; }
                };
                if (r > 0)      accum(hm + c * HSTRIDE + chb,       dC * dinv_of(r - 1, c));
                if (r < WR - 1) accum(hp + c * HSTRIDE + chb,       dC * dinv_of(r + 1, c));
                if (c > 0)      accum(h0 + (c - 1) * HSTRIDE + chb, dC * dinv_of(r, c - 1));
                if (c < WR - 1) accum(h0 + (c + 1) * HSTRIDE + chb, dC * dinv_of(r, c + 1));

                float s1 = 0.f, s2 = 0.f;
#pragma unroll
                for (int j = 0; j < 16; ++j) {
                    acc[j] += lnb[chb + j];
                    s1 += acc[j];
                    s2 += acc[j] * acc[j];
                }
                // 8-lane butterfly: full 128-ch sums live in 8 consecutive lanes
                s1 += __shfl_xor(s1, 1, 32);  s2 += __shfl_xor(s2, 1, 32);
                s1 += __shfl_xor(s1, 2, 32);  s2 += __shfl_xor(s2, 2, 32);
                s1 += __shfl_xor(s1, 4, 32);  s2 += __shfl_xor(s2, 4, 32);
                const float mean = s1 * (1.f / HID);
                const float var  = s2 * (1.f / HID) - mean * mean;
                const float rstd = rsqrtf(var + 1e-5f);

                _Float16* xr = xs + (size_t)(r * WR + c) * XSTRIDE + chb;
#pragma unroll
                for (int j = 0; j < 16; ++j) {
                    float t = (acc[j] - mean) * rstd * lnG[chb + j] + lnB[chb + j];
                    t = fmaxf(t, 0.f) + (float)xr[j];
                    xr[j] = (_Float16)t;
                }
            }
            // no trailing barrier: next GEMM touches only x rows >= r+2 and slot r+2 (&3)
        }
    }

    // ---------------- output: out = x @ Wout + bout ----------------
    __syncthreads();   // protect wb/bo against last stencil row
    for (int idx = tid; idx < 4 * 4 * 32; idx += 256) {   // swizzle Wout (128x64)
        int s  = idx >> 7;
        int l  = idx & 31;
        int gg = l >> 4, mm = l & 15;
        int N  = ((idx >> 5) & 3) * 16 + mm;
        _Float16* dst = wb + (size_t)idx * 16;
#pragma unroll
        for (int e = 0; e < 16; ++e) {
            int K = 32 * s + ((e < 8) ? 8 * gg + e : 16 + 8 * gg + (e - 8));
            dst[e] = (_Float16)Wout[K * OUTD + N];
        }
    }
    for (int ch = tid; ch < OUTD; ch += 256) bo[ch] = bout[ch];
    __syncthreads();

    float* og = out + (size_t)gidx * NNODES * OUTD;
    for (int job = wave; job < 57 * 4; job += 8) {        // wave-uniform
        const int nt = job >> 2;                          // node tile 0..56
        const int ct = job & 3;                           // out-channel tile 0..3
        v8f acc = {};
        const _Float16* arow = xs + (size_t)(nt * 16 + ml) * XSTRIDE;
#pragma unroll
        for (int s = 0; s < 4; ++s) {
            v8h a0 = *(const v8h*)(arow + 32 * s + 8 * gl);
            v8h a1 = *(const v8h*)(arow + 32 * s + 16 + 8 * gl);
            v16h A = __builtin_shufflevector(a0, a1, 0,1,2,3,4,5,6,7,8,9,10,11,12,13,14,15);
            v16h B = *(const v16h*)(wb + (size_t)(((s * 4 + ct) * 32) + lane) * 16);
            acc = __builtin_amdgcn_wmma_f32_16x16x32_f16(false, A, false, B,
                                                         (short)0, acc, false, false);
        }
        const float bb = bo[ct * 16 + ml];
#pragma unroll
        for (int v = 0; v < 8; ++v) {
            int node = nt * 16 + v + 8 * gl;
            if (node < NNODES) og[(size_t)node * OUTD + ct * 16 + ml] = acc[v] + bb;
        }
    }
}

extern "C" void kernel_launch(void* const* d_in, const int* in_sizes, int n_in,
                              void* d_out, int out_size, void* d_ws, size_t ws_size,
                              hipStream_t stream) {
    const int*   grids = (const int*)d_in[0];
    // d_in[1] = edge_index (unused: fixed 30x30 grid topology baked into stencil)
    const float* Win   = (const float*)d_in[2];
    const float* bin_  = (const float*)d_in[3];
    const float* Wg    = (const float*)d_in[4];
    const float* bg    = (const float*)d_in[5];
    const float* gamma = (const float*)d_in[6];
    const float* beta  = (const float*)d_in[7];
    const float* Wout  = (const float*)d_in[8];
    const float* bout  = (const float*)d_in[9];
    float* out = (float*)d_out;

    const int numGrids = in_sizes[0] / NNODES;       // B = 512
    (void)d_ws; (void)ws_size; (void)n_in; (void)out_size;

    (void)hipFuncSetAttribute(reinterpret_cast<const void*>(arc_gnn_encoder_kernel),
                              hipFuncAttributeMaxDynamicSharedMemorySize,
                              (int)SMEM_BYTES);
    arc_gnn_encoder_kernel<<<dim3(numGrids), dim3(256), SMEM_BYTES, stream>>>(
        grids, Win, bin_, Wg, bg, gamma, beta, Wout, bout, out);
}
